// TriFoldLiteATATBA_7078106104184
// MI455X (gfx1250) — matrix operand order repaired and checked by reading, not compiled, not run on hardware
//
#include <hip/hip_runtime.h>
#include <cstdint>
#include <cstddef>

// ---------------- problem constants ----------------
#define LDIM   384
#define SPIX   (LDIM*LDIM)          // 147456 pixels
#define BINSC  63
#define NANCH  32
#define ASTRIDE 12                  // 384 // 32

typedef __attribute__((ext_vector_type(16))) __bf16 v16bf;
typedef __attribute__((ext_vector_type(8)))  __bf16 v8bf;
typedef __attribute__((ext_vector_type(8)))  float  v8f;

__device__ __forceinline__ float dsigmoid(float x){ return 1.f/(1.f+__expf(-x)); }
__device__ __forceinline__ float delu(float x){ return x>0.f ? x : (__expf(x)-1.f); }

// ---------------- f32 -> bf16 (channel pad, Cpad = 1<<csh) ----------------
__global__ __launch_bounds__(256)
void k_to_bf16(const float* __restrict__ in, int Cin,
               __bf16* __restrict__ out, int csh, int n)
{
  int idx = blockIdx.x*blockDim.x + threadIdx.x;
  if (idx >= n) return;
  int c = idx & ((1 << csh) - 1);
  int p = idx >> csh;
  out[idx] = (c < Cin) ? (__bf16)in[(size_t)p*Cin + c] : (__bf16)0.f;
}

// ---------------- weight packers into WMMA B-fragment layout ----------------
// layout: out[((nt*NK + kc)*32 + lane)*16 + e]
//   n = nt*16 + (lane&15);  koff in 32-chunk = (lane>>4)*16 + e
//   (B matrix 32x16: lanes 0-15 hold K=0..15, lanes 16-31 hold K=16..31)
__global__ __launch_bounds__(256)
void k_pack_oihw(const float* __restrict__ w, int Cout, int Cin, int ntaps,
                 int Cinpad, int nTiles, __bf16* __restrict__ out)
{
  const int NK = ntaps * (Cinpad >> 5);
  const int total = nTiles * NK * 512;
  int idx = blockIdx.x*blockDim.x + threadIdx.x;
  if (idx >= total) return;
  int e    = idx & 15;
  int lane = (idx >> 4) & 31;
  int rest = idx >> 9;
  int kc   = rest % NK;
  int nt   = rest / NK;
  const int ncc = Cinpad >> 5;
  int tap = kc / ncc, cc = kc - tap*ncc;
  int n   = nt*16 + (lane & 15);
  int cin = cc*32 + ((lane >> 4) << 4) + e;
  float v = 0.f;
  if (n < Cout && cin < Cin) v = w[((size_t)n*Cin + cin)*ntaps + tap];  // OIHW
  out[idx] = (__bf16)v;
}

__global__ __launch_bounds__(256)
void k_pack_kn(const float* __restrict__ w, int K, int N,
               int Kpad, int nTiles, __bf16* __restrict__ out)
{
  const int NK = Kpad >> 5;
  const int total = nTiles * NK * 512;
  int idx = blockIdx.x*blockDim.x + threadIdx.x;
  if (idx >= total) return;
  int e    = idx & 15;
  int lane = (idx >> 4) & 31;
  int rest = idx >> 9;
  int kc   = rest % NK;
  int nt   = rest / NK;
  int n = nt*16 + (lane & 15);
  int k = kc*32 + ((lane >> 4) << 4) + e;
  float v = 0.f;
  if (n < N && k < K) v = w[(size_t)k*N + n];                            // [K][N]
  out[idx] = (__bf16)v;
}

// ---------------- implicit-GEMM conv / GEMM via WMMA bf16 ----------------
// act: NHWC bf16, channel stride CINPAD = NCC*32. grid = (ceil((W/16)/8), H),
// block = 256 (8 waves); one wave per 16-pixel tile, NTILES*16 out channels.
// Packed weights staged to LDS via GLOBAL_LOAD_ASYNC_TO_LDS (ASYNCcnt).
// actMode: 0 none, 1 relu, 2 sigmoid.
template<int IS3X3, int NCC, int NTILES>
__global__ __launch_bounds__(256)
void k_conv_wmma(const __bf16* __restrict__ act, int H, int W,
                 const __bf16* __restrict__ wpk, int dil,
                 const float* __restrict__ bias, int hasBias,
                 int Cout, int CoutStride,
                 float* __restrict__ out, int actMode)
{
  constexpr int NTAPS  = IS3X3 ? 9 : 1;
  constexpr int NK     = NTAPS * NCC;
  constexpr int CINPAD = NCC * 32;
  __shared__ alignas(16) __bf16 wlds[NTILES * NK * 512];

  // ---- async stage of packed weights into LDS (tracked by ASYNCcnt) ----
  // CH16 is a multiple of 128 -> wave-uniform trip count, EXEC all-1s.
  {
    constexpr int CH16 = NTILES * NK * 64;       // 16-byte chunks
    const uint4* src = (const uint4*)wpk;
    uint4* dst = (uint4*)wlds;
    #pragma unroll 2
    for (int i = threadIdx.x; i < CH16; i += 256) {
      // low 32 bits of the generic pointer == LDS byte offset (ISA 10.2)
      unsigned lofs = (unsigned)(size_t)(dst + i);
      const uint4* gsrc = src + i;
      asm volatile("global_load_async_to_lds_b128 %0, %1, off"
                   :: "v"(lofs), "v"(gsrc) : "memory");
    }
    asm volatile("s_wait_asynccnt 0x0" ::: "memory");
  }
  __syncthreads();

  const int lane = threadIdx.x & 31;
  const int tilesPerRow = W >> 4;
  const int tir = blockIdx.x * 8 + (threadIdx.x >> 5);
  if (tir >= tilesPerRow) return;              // wave-uniform
  const int h  = blockIdx.y;
  const int w0 = tir << 4;

  // prefetch this wave's activation row segment
  __builtin_prefetch(act + ((size_t)h*W + w0)*CINPAD, 0, 3);

  const int m  = lane & 15;                    // A-matrix row (pixel in tile)
  const int kb = (lane >> 4) << 4;             // A k-base within chunk (0 or 16)

  v8f acc[NTILES];
  #pragma unroll
  for (int t = 0; t < NTILES; ++t)
    #pragma unroll
    for (int e = 0; e < 8; ++e) acc[t][e] = 0.f;

  const v16bf* wv = (const v16bf*)wlds;

  #pragma unroll
  for (int kc = 0; kc < NK; ++kc) {
    const int tap = kc / NCC;                  // compile-time after unroll
    const int cc  = kc - tap*NCC;
    int hin = h, win = w0 + m;
    if (IS3X3) {
      const int kh = tap / 3, kw = tap - kh*3;
      hin += (kh-1)*dil;
      win += (kw-1)*dil;
    }
    // A fragment: contiguous k order matching packed B. elements 0..15 -> K kb..kb+15
    // (two contiguous 16B global loads).
    v16bf a;
    if (hin >= 0 && hin < H && win >= 0 && win < W) {
      const __bf16* ap = act + ((size_t)hin*W + win)*CINPAD + cc*32 + kb;
      v8bf lo = *(const v8bf*)ap;
      v8bf hi = *(const v8bf*)(ap + 8);
      #pragma unroll
      for (int e = 0; e < 8; ++e) { a[e] = lo[e]; a[e+8] = hi[e]; }
    } else {
      #pragma unroll
      for (int e = 0; e < 16; ++e) a[e] = (__bf16)0.f;
    }
    #pragma unroll
    for (int nt = 0; nt < NTILES; ++nt) {
      v16bf b = wv[(nt*NK + kc)*32 + lane];    // from LDS
      acc[nt] = __builtin_amdgcn_wmma_f32_16x16x32_bf16(
          false, a, false, b, (short)0, acc[nt], false, false);
    }
  }

  // D layout: element r -> M = r + 8*(lane>>4), N = lane&15 (+16*nt)
  const int mbase = (lane >> 4) << 3;
  #pragma unroll
  for (int nt = 0; nt < NTILES; ++nt) {
    const int co = nt*16 + (lane & 15);
    if (co < Cout) {
      const float bv = hasBias ? bias[co] : 0.f;
      #pragma unroll
      for (int r = 0; r < 8; ++r) {
        float v = acc[nt][r] + bv;
        if (actMode == 1)      v = fmaxf(v, 0.f);
        else if (actMode == 2) v = dsigmoid(v);
        out[((size_t)h*W + w0 + mbase + r)*CoutStride + co] = v;
      }
    }
  }
}

// ---------------- instance norm (64 channels) ----------------
__global__ __launch_bounds__(128)
void k_zero(float* __restrict__ p, int n)
{
  int i = blockIdx.x*blockDim.x + threadIdx.x;
  if (i < n) p[i] = 0.f;
}

__global__ __launch_bounds__(256)
void k_stats64(const float* __restrict__ x, int npix, float* __restrict__ st)
{
  __shared__ float shs[256], shq[256];
  const int c   = threadIdx.x & 63;
  const int sub = threadIdx.x >> 6;
  const int base = blockIdx.x * 1024;
  float s = 0.f, q = 0.f;
  for (int i = 0; i < 256; ++i) {
    int p = base + sub*256 + i;
    if (p < npix) { float v = x[(size_t)p*64 + c]; s += v; q += v*v; }
  }
  shs[threadIdx.x] = s; shq[threadIdx.x] = q;
  __syncthreads();
  if (threadIdx.x < 64) {
    float ts = 0.f, tq = 0.f;
    #pragma unroll
    for (int k2 = 0; k2 < 4; ++k2) { ts += shs[threadIdx.x + 64*k2]; tq += shq[threadIdx.x + 64*k2]; }
    atomicAdd(&st[threadIdx.x],      ts);
    atomicAdd(&st[64 + threadIdx.x], tq);
  }
}

__global__ __launch_bounds__(256)
void k_norm_act(const float* __restrict__ in, const float* __restrict__ st,
                const float* __restrict__ res, float* __restrict__ out,
                int n, int addRes)
{
  int idx = blockIdx.x*blockDim.x + threadIdx.x;
  if (idx >= n) return;
  int c = idx & 63;
  float mu  = st[c] * (1.f/(float)SPIX);
  float var = st[64+c] * (1.f/(float)SPIX) - mu*mu;
  float v = (in[idx] - mu) * rsqrtf(var + 1e-5f);
  if (addRes) v += res[idx];
  out[idx] = delu(v);
}

// ---------------- attention helpers ----------------
__global__ __launch_bounds__(256)
void k_gather(const float* __restrict__ x, __bf16* __restrict__ pl, __bf16* __restrict__ pr)
{
  int idx = blockIdx.x*blockDim.x + threadIdx.x;
  if (idx >= NANCH*LDIM*64) return;
  int c = idx & 63;
  int row = idx >> 6;
  // p_left rows ordered (i, k): pair[i, anchor_k, :]
  int i = row >> 5, k = row & 31;
  pl[idx] = (__bf16)x[((size_t)i*LDIM + k*ASTRIDE)*64 + c];
  // p_right rows ordered (k, j): pair[anchor_k, j, :]
  int k2 = row / LDIM, j = row - k2*LDIM;
  pr[idx] = (__bf16)x[((size_t)k2*ASTRIDE*LDIM + j)*64 + c];
}

__global__ __launch_bounds__(256)
void k_gscalar(const float* __restrict__ td, const float* __restrict__ tq,
               const float* __restrict__ w1, const float* __restrict__ b1,
               const float* __restrict__ w2, const float* __restrict__ b2,
               float* __restrict__ g)
{
  __shared__ float sh[256];
  float cnt = 0.f;
  for (int p = threadIdx.x; p < SPIX; p += 256) cnt += (td[p] > 0.f) ? 1.f : 0.f;
  sh[threadIdx.x] = cnt; __syncthreads();
  for (int s = 128; s > 0; s >>= 1) {
    if (threadIdx.x < s) sh[threadIdx.x] += sh[threadIdx.x + s];
    __syncthreads();
  }
  if (threadIdx.x == 0) {
    float f0 = sh[0] / (float)SPIX;          // coverage
    float f1 = tq[0];                        // template quality
    float f2 = (float)LDIM / 512.f;          // length norm
    float o = b2[0];
    #pragma unroll
    for (int jj = 0; jj < 16; ++jj) {
      float hsum = f0*w1[jj] + f1*w1[16+jj] + f2*w1[32+jj] + b1[jj];
      o += fmaxf(hsum, 0.f) * w2[jj];
    }
    g[0] = dsigmoid(o);
  }
}

// one wave per (i,j); lane = anchor k. scores + template bias + softmax.
__global__ __launch_bounds__(256)
void k_scores(const float* __restrict__ q, const float* __restrict__ lft,
              const float* __restrict__ rgt, const float* __restrict__ td,
              const float* __restrict__ g, float* __restrict__ attn)
{
  int wgid = blockIdx.x * (blockDim.x >> 5) + (threadIdx.x >> 5);
  if (wgid >= SPIX) return;
  const int lane = threadIdx.x & 31;
  const int i = wgid / LDIM, j = wgid - i*LDIM;
  const int anc = lane * ASTRIDE;
  const float* qp = q   + (size_t)wgid*32;
  const float* lp = lft + ((size_t)i*NANCH + lane)*32;
  const float* rp = rgt + ((size_t)lane*LDIM + j)*32;
  float s = 0.f;
  #pragma unroll
  for (int a = 0; a < 32; ++a) {
    float qa = qp[a];
    s += qa*lp[a] + qa*rp[a];
  }
  s *= 0.1767766953f;                                   // 1/sqrt(32)
  float bias = -fabsf(td[(size_t)i*LDIM + anc] + td[(size_t)anc*LDIM + j]
                      - td[wgid]) * (1.f/12.f);
  s += g[0]*bias;
  float mx = s;
  #pragma unroll
  for (int off = 16; off >= 1; off >>= 1) mx = fmaxf(mx, __shfl_xor(mx, off, 32));
  float e = __expf(s - mx);
  float sum = e;
  #pragma unroll
  for (int off = 16; off >= 1; off >>= 1) sum += __shfl_xor(sum, off, 32);
  attn[(size_t)wgid*32 + lane] = e / sum;
}

// update[i,j,a] = sum_k attn * (v_left[i,k,a] + v_right[k,j,a])
__global__ __launch_bounds__(256)
void k_update(const float* __restrict__ attn, const float* __restrict__ vl,
              const float* __restrict__ vr, float* __restrict__ upd)
{
  int idx = blockIdx.x*blockDim.x + threadIdx.x;
  if (idx >= SPIX*32) return;
  int a  = idx & 31;
  int ij = idx >> 5;
  int i = ij / LDIM, j = ij - i*LDIM;
  const float* ap = attn + (size_t)ij*32;
  float s = 0.f;
  #pragma unroll 4
  for (int k = 0; k < 32; ++k) {
    float w = ap[k];
    s += w * vl[((size_t)i*NANCH + k)*32 + a];
    s += w * vr[((size_t)k*LDIM + j)*32 + a];
  }
  upd[idx] = s;
}

__global__ __launch_bounds__(256)
void k_combine(float* __restrict__ x, const float* __restrict__ gate,
               const float* __restrict__ upo, int n)
{
  int idx = blockIdx.x*blockDim.x + threadIdx.x;
  if (idx >= n) return;
  x[idx] += gate[idx] * upo[idx];                       // gate already sigmoid'd
}

// ---------------- heads ----------------
__global__ __launch_bounds__(256)
void k_sym(const float* __restrict__ lg, float* __restrict__ dout,
           __bf16* __restrict__ lbf)
{
  int idx = blockIdx.x*blockDim.x + threadIdx.x;
  if (idx >= SPIX*64) return;
  int c = idx & 63, p = idx >> 6;
  if (c < BINSC) {
    int i = p / LDIM, j = p - i*LDIM;
    float v = 0.5f*(lg[(size_t)p*BINSC + c] + lg[((size_t)j*LDIM + i)*BINSC + c]);
    dout[(size_t)p*BINSC + c] = v;
    lbf[idx] = (__bf16)v;
  } else {
    lbf[idx] = (__bf16)0.f;
  }
}

__global__ __launch_bounds__(256)
void k_conf(const float* __restrict__ c2, const float* __restrict__ w,
            const float* __restrict__ b, float* __restrict__ dout)
{
  int p = blockIdx.x*blockDim.x + threadIdx.x;
  if (p >= SPIX) return;
  float s = b[0];
  #pragma unroll
  for (int c = 0; c < 32; ++c) s += w[c] * c2[(size_t)p*32 + c];
  dout[p] = dsigmoid(s);
}

// ---------------- host-side conv launcher ----------------
static inline int cdiv_h(int a, int b){ return (a + b - 1) / b; }

template<int IS3X3, int NCC, int NTILES>
static void run_conv(hipStream_t stream, __bf16* wpk,
                     const __bf16* act, int H, int W,
                     const float* wsrc, int kn, int dil, int CinK,
                     const float* bias, int Cout, int CoutStride,
                     float* dst, int actMode)
{
  const int ntaps = IS3X3 ? 9 : 1;
  const int NK    = ntaps * NCC;
  const int pkN   = NTILES * NK * 512;
  if (kn) k_pack_kn  <<<cdiv_h(pkN,256),256,0,stream>>>(wsrc, CinK, Cout, NCC*32, NTILES, wpk);
  else    k_pack_oihw<<<cdiv_h(pkN,256),256,0,stream>>>(wsrc, Cout, CinK, ntaps, NCC*32, NTILES, wpk);
  dim3 grid(cdiv_h(W >> 4, 8), H);
  k_conv_wmma<IS3X3,NCC,NTILES><<<grid, 256, 0, stream>>>(
      act, H, W, wpk, dil, bias, bias ? 1 : 0, Cout, CoutStride, dst, actMode);
}

// ---------------- orchestration ----------------
extern "C" void kernel_launch(void* const* d_in, const int* in_sizes, int n_in,
                              void* d_out, int out_size, void* d_ws, size_t ws_size,
                              hipStream_t stream)
{
  const float* features = (const float*)d_in[0];
  const float* td   = (const float*)d_in[1];
  const float* tq   = (const float*)d_in[2];
  const float* in_w = (const float*)d_in[3];
  const float* in_b = (const float*)d_in[4];
  const float* c1w  = (const float*)d_in[5];
  const float* c1b  = (const float*)d_in[6];
  const float* c2w  = (const float*)d_in[7];
  const float* c2b  = (const float*)d_in[8];
  const float* wq   = (const float*)d_in[9];
  const float* wl   = (const float*)d_in[10];
  const float* wr   = (const float*)d_in[11];
  const float* wvl  = (const float*)d_in[12];
  const float* wvr  = (const float*)d_in[13];
  const float* wo   = (const float*)d_in[14];
  const float* wg   = (const float*)d_in[15];
  const float* bg   = (const float*)d_in[16];
  const float* tgw1 = (const float*)d_in[17];
  const float* tgb1 = (const float*)d_in[18];
  const float* tgw2 = (const float*)d_in[19];
  const float* tgb2 = (const float*)d_in[20];
  const float* dw   = (const float*)d_in[21];
  const float* db   = (const float*)d_in[22];
  const float* cf1w = (const float*)d_in[23];
  const float* cf1b = (const float*)d_in[24];
  const float* cf2w = (const float*)d_in[25];
  const float* cf2b = (const float*)d_in[26];
  const float* cf3w = (const float*)d_in[27];
  const float* cf3b = (const float*)d_in[28];
  float* outp = (float*)d_out;

  // ---- workspace carve-out (~190 MB) ----
  char* base = (char*)d_ws;
  size_t off = 0;
  auto alloc = [&](size_t bytes) -> void* {
    void* p = base + off;
    off = (off + bytes + 255) & ~(size_t)255;
    return p;
  };
  float*  xA    = (float*) alloc(sizeof(float)*(size_t)SPIX*64);
  float*  xB    = (float*) alloc(sizeof(float)*(size_t)SPIX*64);
  float*  upo   = (float*) alloc(sizeof(float)*(size_t)SPIX*64);
  float*  qbuf  = (float*) alloc(sizeof(float)*(size_t)SPIX*32);   // q, later reused as `update`
  float*  attnb = (float*) alloc(sizeof(float)*(size_t)SPIX*32);
  __bf16* abf   = (__bf16*)alloc(sizeof(__bf16)*(size_t)SPIX*64);
  __bf16* abf2  = (__bf16*)alloc(sizeof(__bf16)*(size_t)SPIX*32);
  __bf16* plbf  = (__bf16*)alloc(sizeof(__bf16)*(size_t)NANCH*LDIM*64);
  __bf16* prbf  = (__bf16*)alloc(sizeof(__bf16)*(size_t)NANCH*LDIM*64);
  float*  lft   = (float*) alloc(sizeof(float)*(size_t)NANCH*LDIM*32);
  float*  rgt   = (float*) alloc(sizeof(float)*(size_t)NANCH*LDIM*32);
  float*  vlb   = (float*) alloc(sizeof(float)*(size_t)NANCH*LDIM*32);
  float*  vrb   = (float*) alloc(sizeof(float)*(size_t)NANCH*LDIM*32);
  __bf16* wpk   = (__bf16*)alloc(sizeof(__bf16)*4*18*512);
  float*  stats = (float*) alloc(sizeof(float)*256);
  float*  gptr  = stats + 128;

  auto instnorm = [&](float* inb, const float* res, float* dst, int addRes) {
    k_zero<<<1,128,0,stream>>>(stats, 128);
    k_stats64<<<SPIX/1024,256,0,stream>>>(inb, SPIX, stats);
    k_norm_act<<<cdiv_h(SPIX*64,256),256,0,stream>>>(inb, stats, res, dst, SPIX*64, addRes);
  };

  // ---- input 1x1 projection: 41 -> 64 ----
  k_to_bf16<<<cdiv_h(SPIX*64,256),256,0,stream>>>(features, 41, abf, 6, SPIX*64);
  run_conv<0,2,4>(stream, wpk, abf, LDIM, LDIM, in_w, 0, 1, 41, in_b, 64, 64, xA, 0);

  const int DIL[5] = {1, 2, 4, 8, 16};
  for (int i = 0; i < 6; ++i) {
    const int d = DIL[i % 5];
    // conv1 (dilated) -> IN -> elu
    k_to_bf16<<<cdiv_h(SPIX*64,256),256,0,stream>>>(xA, 64, abf, 6, SPIX*64);
    run_conv<1,2,4>(stream, wpk, abf, LDIM, LDIM, c1w + (size_t)i*36864, 0, d, 64,
                    c1b + i*64, 64, 64, xB, 0);
    instnorm(xB, nullptr, xB, 0);
    // conv2 -> IN -> +res -> elu
    k_to_bf16<<<cdiv_h(SPIX*64,256),256,0,stream>>>(xB, 64, abf, 6, SPIX*64);
    run_conv<1,2,4>(stream, wpk, abf, LDIM, LDIM, c2w + (size_t)i*36864, 0, 1, 64,
                    c2b + i*64, 64, 64, xB, 0);
    instnorm(xB, xA, xA, 1);

    if (i & 1) {   // anchor attention after blocks 2, 4, 6
      k_to_bf16<<<cdiv_h(SPIX*64,256),256,0,stream>>>(xA, 64, abf, 6, SPIX*64);
      run_conv<0,2,2>(stream, wpk, abf, LDIM, LDIM, wq, 1, 1, 64, nullptr, 32, 32, qbuf, 0);
      k_gather<<<cdiv_h(NANCH*LDIM*64,256),256,0,stream>>>(xA, plbf, prbf);
      // 12288 anchor-rows reshaped as H=48, W=256 for full 2-D grid occupancy
      run_conv<0,2,2>(stream, wpk, plbf, 48, 256, wl,  1, 1, 64, nullptr, 32, 32, lft, 0);
      run_conv<0,2,2>(stream, wpk, prbf, 48, 256, wr,  1, 1, 64, nullptr, 32, 32, rgt, 0);
      run_conv<0,2,2>(stream, wpk, plbf, 48, 256, wvl, 1, 1, 64, nullptr, 32, 32, vlb, 0);
      run_conv<0,2,2>(stream, wpk, prbf, 48, 256, wvr, 1, 1, 64, nullptr, 32, 32, vrb, 0);
      k_gscalar<<<1,256,0,stream>>>(td, tq, tgw1, tgb1, tgw2, tgb2, gptr);
      k_scores<<<cdiv_h(SPIX,8),256,0,stream>>>(qbuf, lft, rgt, td, gptr, attnb);
      k_update<<<cdiv_h(SPIX*32,256),256,0,stream>>>(attnb, vlb, vrb, qbuf);   // qbuf := update
      k_to_bf16<<<cdiv_h(SPIX*32,256),256,0,stream>>>(qbuf, 32, abf2, 5, SPIX*32);
      run_conv<0,1,4>(stream, wpk, abf2, LDIM, LDIM, wo, 1, 1, 32, nullptr, 64, 64, upo, 0);
      run_conv<0,2,4>(stream, wpk, abf,  LDIM, LDIM, wg, 1, 1, 64, bg,      64, 64, xB,  2);
      k_combine<<<cdiv_h(SPIX*64,256),256,0,stream>>>(xA, xB, upo, SPIX*64);
    }
  }

  // ---- distance head: 3x3 64->63, symmetrize ----
  k_to_bf16<<<cdiv_h(SPIX*64,256),256,0,stream>>>(xA, 64, abf, 6, SPIX*64);
  run_conv<1,2,4>(stream, wpk, abf, LDIM, LDIM, dw, 0, 1, 64, db, BINSC, BINSC, xB, 0);
  k_sym<<<cdiv_h(SPIX*64,256),256,0,stream>>>(xB, outp, abf);  // d_out logits + bf16 for cf head

  // ---- confidence head ----
  run_conv<1,2,2>(stream, wpk, abf, LDIM, LDIM, cf1w, 0, 1, BINSC, cf1b, 32, 32, xA, 1);
  k_to_bf16<<<cdiv_h(SPIX*32,256),256,0,stream>>>(xA, 32, abf2, 5, SPIX*32);
  run_conv<1,1,2>(stream, wpk, abf2, LDIM, LDIM, cf2w, 0, 1, 32, cf2b, 32, 32, xB, 1);
  k_conf<<<cdiv_h(SPIX,256),256,0,stream>>>(xB, cf3w, cf3b, outp + (size_t)SPIX*BINSC);

  (void)in_sizes; (void)n_in; (void)out_size; (void)ws_size;
}
